// BinaryConv2d_55113020342317
// MI455X (gfx1250) — matrix operand order repaired and checked by reading, not compiled, not run on hardware
//
#include <hip/hip_runtime.h>
#include <stdint.h>

typedef __attribute__((ext_vector_type(16))) __bf16   v16bf;
typedef __attribute__((ext_vector_type(8)))  float    v8f;
typedef __attribute__((ext_vector_type(4)))  uint32_t u32x4;
typedef __attribute__((ext_vector_type(8)))  int      i32x8;
typedef __attribute__((ext_vector_type(4)))  int      i32x4;

#define TILE_M   128
#define TILE_N   64
#define TILE_K   32
#define K_TOTAL  1152         // 128 * 9
#define K_STEPS  36           // 1152 / 32
#define A_STRIDE 40           // bf16 elems per LDS row (32 data + 8 pad, 16B-aligned chunks)
#define B_STRIDE 40           // matches TDM pad: 64B data + 16B pad per row
#define C_STRIDE 65           // f32 elems per LDS row in epilogue (bank-conflict pad)
#define HW       3136         // 56*56

#define A_BUF_BYTES (TILE_M * A_STRIDE * 2)   // 10240
#define B_BUF_BYTES (TILE_N * B_STRIDE * 2)   //  5120

// ---- fp32 -> bf16 round-to-nearest-even ----
__device__ __forceinline__ __bf16 f32_to_bf16(float f) {
  union { float f; uint32_t u; } in; in.f = f;
  uint32_t u = in.u;
  u += 0x7FFFu + ((u >> 16) & 1u);
  union { uint16_t s; __bf16 b; } out; out.s = (uint16_t)(u >> 16);
  return out.b;
}

// ---- Kernel 0: binarize + relayout weights: Wtb[o][(ky*3+kx)*128 + ci] = bf16(sign(W[o][ci][ky][kx])) ----
__global__ void wprep_kernel(const float* __restrict__ W, __bf16* __restrict__ Wtb) {
  int t = blockIdx.x * blockDim.x + threadIdx.x;
  if (t >= 256 * K_TOTAL) return;
  int o  = t / K_TOTAL;
  int r  = t % K_TOTAL;
  int s  = r >> 7;        // tap index 0..8
  int ci = r & 127;
  int ky = s / 3, kx = s % 3;
  float w = W[(((size_t)o * 128 + ci) * 3 + ky) * 3 + kx];
  float sg = (w > 0.f) ? 1.f : ((w < 0.f) ? -1.f : 0.f);
  Wtb[t] = f32_to_bf16(sg);
}

// ---- TDM: load 64x32 bf16 weight tile (row stride K_TOTAL) into LDS with +16B/row padding ----
__device__ __forceinline__ void tdm_load_B(const __bf16* gsrc, uint32_t lds_addr) {
  uint64_t ga = (uint64_t)(uintptr_t)gsrc;
  u32x4 g0;
  g0.x = 1u;                                            // count=1 (valid), user mode
  g0.y = lds_addr;                                      // LDS byte address
  g0.z = (uint32_t)ga;                                  // global_addr[31:0]
  g0.w = (uint32_t)((ga >> 32) & 0x01FFFFFFu)           // global_addr[56:32]
         | 0x80000000u;                                 // type=2 ("image")
  i32x8 g1;
  g1[0] = (int)((1u << 16)      // data_size = 2 bytes
              | (1u << 20)      // pad_enable
              | (3u << 22)      // pad_interval: 16 DWORDs (= 64B = one 32-elem row)
              | (3u << 25));    // pad_amount: 4 DWORDs (= 16B) -> LDS row stride 80B
  g1[1] = (int)(((uint32_t)K_TOTAL & 0xFFFFu) << 16);   // tensor_dim0[15:0]
  g1[2] = (int)(((uint32_t)K_TOTAL >> 16) | (256u << 16)); // tensor_dim0[31:16] | tensor_dim1[15:0]
  g1[3] = (int)((256u >> 16) | ((uint32_t)TILE_K << 16));  // tensor_dim1[31:16] | tile_dim0
  g1[4] = TILE_N;                                       // tile_dim1 = 64 rows (tile_dim2 = 0)
  g1[5] = K_TOTAL;                                      // tensor_dim0_stride[31:0]
  g1[6] = 0;
  g1[7] = 0;
  i32x4 z4 = {0, 0, 0, 0};                              // 2D tensor: groups 2/3 unused
  i32x8 z8 = {0, 0, 0, 0, 0, 0, 0, 0};                  // 6-arg toolchain extra group
  __builtin_amdgcn_tensor_load_to_lds(g0, g1, z4, z4, z8, 0);
}

// ---- stage A: gather + bf16-convert one 128x32 activation slice into LDS ----
__device__ __forceinline__ void stage_A(const float* __restrict__ x, __bf16* Abuf,
                                        int ks, long xoff_img, int oy, int ox,
                                        int mA, int ciHalf) {
  const int kk  = ks * TILE_K;
  const int s   = kk >> 7;          // filter tap 0..8 (fixed within K-slice)
  const int ci0 = kk & 127;
  const int ky  = s / 3, kx = s % 3;
  const int iy  = oy + ky - 1;
  const int ix  = ox + kx - 1;
  const bool inb = (iy >= 0) & (iy < 56) & (ix >= 0) & (ix < 56);

  __align__(16) __bf16 abuf[16];
  const float* xp = x + xoff_img + (long)(ci0 + ciHalf) * HW + (iy * 56 + ix);
  #pragma unroll
  for (int j = 0; j < 16; ++j) {
    float v = inb ? xp[(long)j * HW] : 0.f;
    abuf[j] = f32_to_bf16(v);
  }
  uint4* dstA = (uint4*)&Abuf[mA * A_STRIDE + ciHalf];
  dstA[0] = ((const uint4*)abuf)[0];
  dstA[1] = ((const uint4*)abuf)[1];
}

// ---- Kernel 1: implicit-GEMM binary conv, double-buffered LDS, TDM B-loads, bf16 WMMA ----
__global__ __launch_bounds__(256)
void bconv_wmma_kernel(const float* __restrict__ x,
                       const __bf16* __restrict__ Wtb,
                       float* __restrict__ out) {
  __shared__ __align__(16) char smem[TILE_M * C_STRIDE * 4];  // 33280 B (C overlay is largest)

  // scalar locals only: pointer *arrays* here get lowered to static initializers
  // with addrspacecast(LDS->generic), which ld.lld rejects.
  __bf16* A0 = (__bf16*)(smem);
  __bf16* A1 = (__bf16*)(smem + A_BUF_BYTES);
  __bf16* B0 = (__bf16*)(smem + 2 * A_BUF_BYTES);
  __bf16* B1 = (__bf16*)(smem + 2 * A_BUF_BYTES + B_BUF_BYTES);
  float*  Csub = (float*)(smem);

  const int tid  = threadIdx.x;
  const int lane = tid & 31;
  const int wv   = tid >> 5;          // wave 0..7 -> 16-row M strip

  const int mtile = blockIdx.x;       // 0..783
  const int cout0 = blockIdx.y * TILE_N;
  const __bf16* Wrow = Wtb + (size_t)cout0 * K_TOTAL;

  // Per-thread A-load coordinates: m fixed per thread, 16 ci values
  const int mA     = tid & 127;
  const int ciHalf = (tid >> 7) * 16;         // 0 or 16
  const int gm  = mtile * TILE_M + mA;
  const int img = gm / HW;
  const int pix = gm % HW;
  const int oy  = pix / 56;
  const int ox  = pix % 56;
  const long xoff_img = (long)img * 128 * HW;

  const uint32_t ldsB0 = (uint32_t)(uintptr_t)(void*)B0;
  const uint32_t ldsB1 = (uint32_t)(uintptr_t)(void*)B1;

  v8f acc[4] = { v8f{}, v8f{}, v8f{}, v8f{} };

  // ---- prologue: stage slice 0 into buffer 0 ----
  stage_A(x, A0, 0, xoff_img, oy, ox, mA, ciHalf);
  if (wv == 0) tdm_load_B(Wrow, ldsB0);
  __builtin_amdgcn_s_wait_tensorcnt(0);
  __syncthreads();

  for (int ks = 0; ks < K_STEPS; ++ks) {
    const int cur = ks & 1;
    __bf16* Acur = cur ? A1 : A0;
    __bf16* Bcur = cur ? B1 : B0;

    // ---- overlap: stage slice ks+1 while computing slice ks ----
    if (ks + 1 < K_STEPS) {
      __bf16*  Anxt   = cur ? A0 : A1;
      uint32_t ldsBnx = cur ? ldsB0 : ldsB1;
      stage_A(x, Anxt, ks + 1, xoff_img, oy, ox, mA, ciHalf);
      if (wv == 0) tdm_load_B(Wrow + (size_t)(ks + 1) * TILE_K, ldsBnx);
    }

    // --- A fragment (16x32 bf16 ISA layout): lanes 0-15: K0-7,K16-23 ; lanes 16-31: K8-15,K24-31 ---
    union { v16bf v; uint4 u[2]; } afrag;
    {
      const int r  = wv * 16 + (lane & 15);
      const int kb = (lane < 16) ? 0 : 8;
      const __bf16* ap = &Acur[r * A_STRIDE];
      afrag.u[0] = *(const uint4*)(ap + kb);
      afrag.u[1] = *(const uint4*)(ap + 16 + kb);
    }
    // --- B fragments (32x16): lane holds one column; lanes 0-15: K0-7,K16-23 ; lanes 16-31: K8-15,K24-31 ---
    const int cb   = lane & 15;
    const int koff = (lane < 16) ? 0 : 8;
    #pragma unroll
    for (int ni = 0; ni < 4; ++ni) {
      union { v16bf v; uint4 u[2]; } bfrag;
      const __bf16* bp = &Bcur[(ni * 16 + cb) * B_STRIDE + koff];
      bfrag.u[0] = ((const uint4*)bp)[0];
      bfrag.u[1] = ((const uint4*)bp)[2];
      acc[ni] = __builtin_amdgcn_wmma_f32_16x16x32_bf16(
          /*neg_a=*/false, afrag.v, /*neg_b=*/false, bfrag.v,
          /*c_mod=*/(short)0, acc[ni], /*reuse_a=*/false, /*reuse_b=*/false);
    }

    __builtin_amdgcn_s_wait_tensorcnt(0);   // wave0's TDM for slice ks+1 complete
    __syncthreads();                        // publish next buffers / retire current
  }

  // ---- epilogue: registers -> LDS (transpose) -> coalesced NCHW global stores ----
  #pragma unroll
  for (int ni = 0; ni < 4; ++ni) {
    const int m = wv * 16 + (lane >> 4) * 8;    // + j below
    const int n = ni * 16 + (lane & 15);
    #pragma unroll
    for (int j = 0; j < 8; ++j)
      Csub[(m + j) * C_STRIDE + n] = acc[ni][j];
  }
  __syncthreads();

  #pragma unroll
  for (int it = 0; it < 32; ++it) {
    const int flat   = it * 256 + tid;          // 0..8191 over 64(cout) x 128(m)
    const int cout_l = flat >> 7;
    const int m_l    = flat & 127;
    const int g  = mtile * TILE_M + m_l;
    const int im = g / HW;
    const int px = g % HW;
    out[((size_t)(im * 256 + cout0 + cout_l)) * HW + px] = Csub[m_l * C_STRIDE + cout_l];
  }
}

extern "C" void kernel_launch(void* const* d_in, const int* in_sizes, int n_in,
                              void* d_out, int out_size, void* d_ws, size_t ws_size,
                              hipStream_t stream) {
  (void)in_sizes; (void)n_in; (void)out_size; (void)ws_size;
  const float* x = (const float*)d_in[0];
  const float* W = (const float*)d_in[1];
  float* out = (float*)d_out;
  __bf16* Wtb = (__bf16*)d_ws;                 // 256*1152 bf16 = 576 KB scratch

  wprep_kernel<<<(256 * K_TOTAL + 255) / 256, 256, 0, stream>>>(W, Wtb);

  dim3 grid(784, 4);                           // 100352/128 M-tiles x 256/64 N-tiles
  bconv_wmma_kernel<<<grid, 256, 0, stream>>>(x, Wtb, out);
}
